// MSTGCN_block_83708912599823
// MI455X (gfx1250) — matrix operand order repaired and checked by reading, not compiled, not run on hardware
//
#include <hip/hip_runtime.h>
#include <hip/hip_bf16.h>

typedef _Float16 half_t;
typedef __attribute__((ext_vector_type(16))) _Float16 v16h;
typedef __attribute__((ext_vector_type(8)))  _Float16 v8h;
typedef __attribute__((ext_vector_type(2)))  _Float16 v2h;
typedef __attribute__((ext_vector_type(8)))  float    v8f;

constexpr int N_ = 8, T_ = 32, V_ = 64, F_ = 64, K_ = 3, CF_ = 64, TF_ = 64, KT_ = 3;
constexpr float ALPHA_ = 1e-4f, LN_EPS_ = 1e-5f;

// ---------------- CDNA5 async global->LDS copy (ASYNCcnt-tracked DMA path) ------------
__device__ inline void async_ld_b128(unsigned lds_off, const void* gaddr) {
  asm volatile("global_load_async_to_lds_b128 %0, %1, off"
               :: "v"(lds_off), "v"(gaddr) : "memory");
}
__device__ inline void wait_async0() {
  asm volatile("s_wait_asynccnt 0" ::: "memory");
}

// ---------------- WMMA fragment helpers (CDNA5 16x16x32 f16, wave32) ----------------
// A fragment from row-major A[m][k]: lane L -> m = mBase + L%16 ;
// elems e -> k = kBase + 16*(L>>4) + e : 16 contiguous halves.
__device__ inline v16h ld_a_rm(const half_t* src, int ld, int mBase, int kBase, int lane) {
  int m = mBase + (lane & 15);
  int kb = kBase + ((lane >> 4) << 4);
  const v8h* p = (const v8h*)(src + m * ld + kb);
  v8h lo = p[0];
  v8h hi = p[1];
  return __builtin_shufflevector(lo, hi, 0, 1, 2, 3, 4, 5, 6, 7, 8, 9, 10, 11, 12, 13, 14, 15);
}
// B fragment from column-major B stored as srcT[n][k]:
// lane L -> n = nBase + L%16 ; elems e -> k = kBase + 16*(L>>4) + e : 16 contiguous halves.
__device__ inline v16h ld_b_cm(const half_t* srcT, int ld, int kBase, int nBase, int lane) {
  int n = nBase + (lane & 15);
  int kb = kBase + ((lane >> 4) << 4);
  const v8h* p = (const v8h*)(srcT + n * ld + kb);
  v8h lo = p[0];
  v8h hi = p[1];
  return __builtin_shufflevector(lo, hi, 0, 1, 2, 3, 4, 5, 6, 7, 8, 9, 10, 11, 12, 13, 14, 15);
}
__device__ inline v8f wmma_f16(v16h a, v16h b, v8f c) {
  return __builtin_amdgcn_wmma_f32_16x16x32_f16(false, a, false, b, (short)0, c, false, false);
}

// ---------------- prep: convert + transpose weights ----------------
// ThGLhT/ThSDhT: [k][o][f] ; WghT/WshT: [dt][o][c] ; chebpT: [k][j][i] (f32)
__global__ void k_prep(const float* ThGL, const float* ThSD, const float* Wg, const float* Ws,
                       const float* chebp, half_t* ThGLhT, half_t* ThSDhT,
                       half_t* WghT, half_t* WshT, float* chebpT) {
  int i = blockIdx.x * blockDim.x + threadIdx.x;
  if (i < K_ * F_ * CF_) {
    int k = i / (F_ * CF_), f = (i / CF_) % F_, o = i % CF_;
    ThGLhT[((size_t)k * CF_ + o) * F_ + f] = (half_t)ThGL[i];
    ThSDhT[((size_t)k * CF_ + o) * F_ + f] = (half_t)ThSD[i];
  }
  if (i < TF_ * CF_ * KT_) {  // W (TF,CF,KT,1)
    int o = i / (CF_ * KT_), c = (i / KT_) % CF_, dt = i % KT_;
    WghT[((size_t)dt * TF_ + o) * CF_ + c] = (half_t)Wg[i];
    WshT[((size_t)dt * TF_ + o) * CF_ + c] = (half_t)Ws[i];
  }
  if (i < K_ * V_ * V_) {
    int k = i / (V_ * V_), r = (i / V_) % V_, c = i % V_;
    chebpT[((size_t)k * V_ + c) * V_ + r] = chebp[i];
  }
}

// xhT[n,t,f,v] = (half)x[n,t,v,f]  (tiled transpose, coalesced in and out)
__global__ __launch_bounds__(256) void k_x2hT(const float* x, half_t* xhT) {
  int nt = blockIdx.x;
  __shared__ __attribute__((aligned(16))) half_t sx[V_ * F_];
  const float* xp = x + (size_t)nt * V_ * F_;
  for (int idx = threadIdx.x; idx < V_ * F_; idx += 256) sx[idx] = (half_t)xp[idx];
  __syncthreads();
  half_t* op = xhT + (size_t)nt * V_ * F_;
  for (int idx = threadIdx.x; idx < V_ * F_; idx += 256) {
    int f = idx >> 6, v = idx & 63;
    op[idx] = sx[v * F_ + f];
  }
}

// ---------------- temporal attention: TAtt (N,T,T) ----------------
__global__ __launch_bounds__(256) void k_tatt(const float* x, const float* U1, const float* U2,
    const float* U3, const float* b_e, const float* V_e, float* TAtt) {
  int n = blockIdx.x, tid = threadIdx.x;
  __shared__ float lhs1[T_ * F_];
  __shared__ float rhs[V_ * T_];
  __shared__ float lhs[T_ * V_];
  __shared__ float sig[T_ * T_];
  __shared__ float Em[T_ * T_];
  const float* xn = x + (size_t)n * T_ * V_ * F_;
  for (int idx = tid; idx < T_ * F_; idx += 256) {   // lhs1[t,f] = sum_v x*U1
    int t = idx / F_, f = idx % F_;
    float s = 0.f;
    for (int v = 0; v < V_; ++v) s += xn[(t * V_ + v) * F_ + f] * U1[v];
    lhs1[idx] = s;
  }
  for (int idx = tid; idx < V_ * T_; idx += 256) {   // rhs[v,t] = sum_f x*U3
    int v = idx / T_, t = idx % T_;
    const float* xr = xn + (t * V_ + v) * F_;
    float s = 0.f;
    for (int f = 0; f < F_; ++f) s += xr[f] * U3[f];
    rhs[idx] = s;
  }
  __syncthreads();
  for (int idx = tid; idx < T_ * V_; idx += 256) {   // lhs = lhs1 @ U2
    int t = idx / V_, v = idx % V_;
    float s = 0.f;
    for (int f = 0; f < F_; ++f) s += lhs1[t * F_ + f] * U2[f * V_ + v];
    lhs[idx] = s;
  }
  __syncthreads();
  for (int idx = tid; idx < T_ * T_; idx += 256) {   // sigmoid(lhs@rhs + b_e)
    int t = idx / T_, s2 = idx % T_;
    float s = 0.f;
    for (int v = 0; v < V_; ++v) s += lhs[t * V_ + v] * rhs[v * T_ + s2];
    s += b_e[idx];
    sig[idx] = 1.f / (1.f + expf(-s));
  }
  __syncthreads();
  for (int idx = tid; idx < T_ * T_; idx += 256) {   // Em[a,i] = sum_b sig[a,b]*V_e[i,b]
    int a = idx / T_, i = idx % T_;
    float s = 0.f;
    for (int b = 0; b < T_; ++b) s += sig[a * T_ + b] * V_e[i * T_ + b];
    Em[idx] = s;
  }
  __syncthreads();
  if (tid < T_) {                                    // softmax over axis=1 (rows a), per column i
    int i = tid;
    float mx = -1e30f;
    for (int a = 0; a < T_; ++a) mx = fmaxf(mx, Em[a * T_ + i]);
    float sum = 0.f;
    for (int a = 0; a < T_; ++a) { float e = expf(Em[a * T_ + i] - mx); Em[a * T_ + i] = e; sum += e; }
    for (int a = 0; a < T_; ++a) TAtt[((size_t)n * T_ + a) * T_ + i] = Em[a * T_ + i] / sum;
  }
}

// ---------------- x_TAtt[n,t,v,f] = sum_s x[n,s,v,f]*TAtt[n,s,t] ----------------
__global__ __launch_bounds__(256) void k_xtatt(const float* x, const float* TAtt, float* xT) {
  int nt = blockIdx.x, n = nt / T_, t = nt % T_;
  __shared__ float w[T_];
  if (threadIdx.x < T_) w[threadIdx.x] = TAtt[((size_t)n * T_ + threadIdx.x) * T_ + t];
  __syncthreads();
  const float* xn = x + (size_t)n * T_ * V_ * F_;
  float* o = xT + (size_t)nt * V_ * F_;
  for (int idx = threadIdx.x; idx < V_ * F_; idx += 256) {
    float s = 0.f;
    for (int s2 = 0; s2 < T_; ++s2) s += xn[(size_t)s2 * V_ * F_ + idx] * w[s2];
    o[idx] = s;
  }
}

// ---------------- spatial attention: SAtt (N,V,V) ----------------
__global__ __launch_bounds__(256) void k_satt(const float* xT, const float* W1, const float* W2,
    const float* W3, const float* b_s, const float* V_s, float* SAtt) {
  int n = blockIdx.x, tid = threadIdx.x;
  __shared__ float buf1[V_ * F_];  // lhs1 then Em
  __shared__ float lhs[V_ * T_];
  __shared__ float rhs[T_ * V_];
  __shared__ float sig[V_ * V_];
  const float* xn = xT + (size_t)n * T_ * V_ * F_;
  for (int idx = tid; idx < V_ * F_; idx += 256) {   // lhs1[v,f] = sum_t xT*W1
    int v = idx / F_, f = idx % F_;
    float s = 0.f;
    for (int t = 0; t < T_; ++t) s += xn[(t * V_ + v) * F_ + f] * W1[t];
    buf1[idx] = s;
  }
  for (int idx = tid; idx < T_ * V_; idx += 256) {   // rhs[t,v] = sum_f xT*W3
    int t = idx / V_, v = idx % V_;
    const float* xr = xn + (t * V_ + v) * F_;
    float s = 0.f;
    for (int f = 0; f < F_; ++f) s += xr[f] * W3[f];
    rhs[idx] = s;
  }
  __syncthreads();
  for (int idx = tid; idx < V_ * T_; idx += 256) {   // lhs = lhs1 @ W2
    int v = idx / T_, t = idx % T_;
    float s = 0.f;
    for (int f = 0; f < F_; ++f) s += buf1[v * F_ + f] * W2[f * T_ + t];
    lhs[idx] = s;
  }
  __syncthreads();
  for (int idx = tid; idx < V_ * V_; idx += 256) {   // sigmoid(lhs@rhs + b_s)
    int v = idx >> 6, u = idx & 63;
    float s = 0.f;
    for (int t = 0; t < T_; ++t) s += lhs[v * T_ + t] * rhs[t * V_ + u];
    s += b_s[idx];
    sig[idx] = 1.f / (1.f + expf(-s));
  }
  __syncthreads();
  for (int idx = tid; idx < V_ * V_; idx += 256) {   // Em[a,i] = sum_b sig[a,b]*V_s[i,b]
    int a = idx >> 6, i = idx & 63;
    float s = 0.f;
    for (int b = 0; b < V_; ++b) s += sig[a * V_ + b] * V_s[i * V_ + b];
    buf1[idx] = s;
  }
  __syncthreads();
  if (tid < V_) {                                    // softmax over rows a, per column i
    int i = tid;
    float mx = -1e30f;
    for (int a = 0; a < V_; ++a) mx = fmaxf(mx, buf1[a * V_ + i]);
    float sum = 0.f;
    for (int a = 0; a < V_; ++a) { float e = expf(buf1[a * V_ + i] - mx); buf1[a * V_ + i] = e; sum += e; }
    for (int a = 0; a < V_; ++a) SAtt[((size_t)n * V_ + a) * V_ + i] = buf1[a * V_ + i] / sum;
  }
}

// ---------------- graph learn: S (N,T,V,V), column-normalized over i ----------------
__global__ __launch_bounds__(256) void k_S(const float* x, const float* a_gl, float* S) {
  int nt = blockIdx.x, tid = threadIdx.x;
  __shared__ __attribute__((aligned(16))) float sx[V_ * F_];
  __shared__ float tS[V_ * V_];
  __shared__ float sa[F_];
  __shared__ float csum[V_];
  const float* xp = x + (size_t)nt * V_ * F_;
  for (int i4 = tid; i4 < V_ * F_ / 4; i4 += 256) ((float4*)sx)[i4] = ((const float4*)xp)[i4];
  if (tid < F_) sa[tid] = a_gl[tid];
  __syncthreads();
  for (int idx = tid; idx < V_ * V_; idx += 256) {
    int i = idx >> 6, j = idx & 63;
    float s = 0.f;
    for (int f = 0; f < F_; ++f) s += fabsf(sx[i * F_ + f] - sx[j * F_ + f]) * sa[f];
    tS[idx] = expf(s * (1.f / F_));
  }
  __syncthreads();
  if (tid < V_) {
    float s = 0.f;
    for (int i = 0; i < V_; ++i) s += tS[i * V_ + tid];
    csum[tid] = s + 1e-8f;
  }
  __syncthreads();
  float* Sp = S + (size_t)nt * V_ * V_;
  for (int idx = tid; idx < V_ * V_; idx += 256) Sp[idx] = tS[idx] / csum[idx & 63];
}

__global__ void k_smean(const float* S, float* Smean) {
  int i = blockIdx.x * 256 + threadIdx.x;
  if (i < T_ * V_ * V_) {
    float s = 0.f;
    for (int nn = 0; nn < N_; ++nn) s += S[(size_t)nn * T_ * V_ * V_ + i];
    Smean[i] = s * (1.f / N_);
  }
}

// D2[i,j] = sum_f ( (1/(N*T)) * sum_{n,t} |x[n,t,i,f]-x[n,t,j,f]| )^2
__global__ __launch_bounds__(64) void k_d2(const float* x, float* D2) {
  int i = blockIdx.x >> 6, j = blockIdx.x & 63, f = threadIdx.x;
  float s = 0.f;
  for (int nt = 0; nt < N_ * T_; ++nt) {
    const float* base = x + (size_t)nt * V_ * F_;
    s += fabsf(base[i * F_ + f] - base[j * F_ + f]);
  }
  s *= (1.f / (N_ * T_));
  s = s * s;
  __shared__ float red[64];
  red[f] = s;
  __syncthreads();
  if (f == 0) {
    float t = 0.f;
    for (int q = 0; q < 64; ++q) t += red[q];
    D2[blockIdx.x] = t;
  }
}

__global__ __launch_bounds__(256) void k_loss(const float* Smean, const float* D2, float* out, int outMain) {
  __shared__ float r1[256], r2[256];
  int tid = threadIdx.x;
  float a1 = 0.f, a2 = 0.f;
  for (int idx = tid; idx < T_ * V_ * V_; idx += 256) {
    float sm = Smean[idx];
    a1 += sm * D2[idx & (V_ * V_ - 1)];
    a2 += sm * sm;
  }
  r1[tid] = a1; r2[tid] = a2;
  __syncthreads();
  for (int s = 128; s > 0; s >>= 1) {
    if (tid < s) { r1[tid] += r1[tid + s]; r2[tid] += r2[tid + s]; }
    __syncthreads();
  }
  if (tid == 0) { out[outMain] = ALPHA_ * r1[0]; out[outMain + 1] = ALPHA_ * r2[0]; }
}

// ---------------- Chebyshev GL branch: fused Laplacian + WMMA chain ----------------
// All matmuls in transposed orientation so every C-fragment store is contiguous.
__global__ __launch_bounds__(128) void k_cheb_gl(const float* S, const float* SAtt,
    const half_t* xhT, const half_t* ThT, half_t* hOut) {
  int nt = blockIdx.x, n = nt / T_, tid = threadIdx.x;
  __shared__ __attribute__((aligned(16))) float  sL[V_ * V_];
  __shared__ float  sd[V_];
  __shared__ __attribute__((aligned(16))) half_t sAttT[V_ * V_];  // Att^T: [j][i]
  __shared__ __attribute__((aligned(16))) half_t sTkT[V_ * V_];   // T_k^T: [j][i] (B col-major)
  __shared__ __attribute__((aligned(16))) half_t sXT[V_ * V_];    // x^T: [f][v] (A row-major)
  __shared__ __attribute__((aligned(16))) half_t sZ[V_ * V_];     // Z: [j][f]
  const float* Sp = S + (size_t)nt * V_ * V_;
  const float* Attp = SAtt + (size_t)n * V_ * V_;
  const half_t* xpT = xhT + (size_t)nt * V_ * F_;
  __builtin_prefetch(ThT, 0, 1);
  // async DMA of x^T tile into LDS; overlapped with the Laplacian build below
  unsigned ldsX = (unsigned)(uintptr_t)sXT;
  for (int i8 = tid; i8 < V_ * F_ / 8; i8 += 128)
    async_ld_b128(ldsX + i8 * 16, (const void*)(xpT + i8 * 8));
  for (int i4 = tid; i4 < V_ * V_ / 4; i4 += 128) ((float4*)sL)[i4] = ((const float4*)Sp)[i4];
  for (int idx = tid; idx < V_ * V_; idx += 128) {   // Att^T staging
    int i = idx >> 6, j = idx & 63;
    sAttT[j * V_ + i] = (half_t)Attp[idx];
  }
  __syncthreads();
  for (int idx = tid; idx < V_ * V_; idx += 128) {   // Sm = min(S, S^T), pair-owned in place
    int i = idx >> 6, j = idx & 63;
    if (i <= j) {
      float a = sL[i * V_ + j], b = sL[j * V_ + i];
      float mn = fminf(a, b);
      sL[i * V_ + j] = mn; sL[j * V_ + i] = mn;
    }
  }
  __syncthreads();
  if (tid < V_) {                                    // d[j] = sum_i Sm[i,j]
    float s = 0.f;
    for (int i = 0; i < V_; ++i) s += sL[i * V_ + tid];
    sd[tid] = s;
  }
  __syncthreads();
  for (int idx = tid; idx < V_ * V_; idx += 128) {   // L = D - Sm - I  (symmetric)
    int i = idx >> 6, j = idx & 63;
    sL[idx] = ((i == j) ? (sd[j] - 1.f) : 0.f) - sL[idx];
  }
  wait_async0();                                     // x^T DMA done (own wave) ...
  __syncthreads();                                   // ... and visible block-wide
  int wave = tid >> 5, lane = tid & 31;
  // loop-invariant A fragments of X^T (sXT is never rewritten)
  v16h xa0 = ld_a_rm(sXT, V_, 16 * wave, 0, lane);
  v16h xa1 = ld_a_rm(sXT, V_, 16 * wave, 32, lane);
  v8f vz = {};
  v8f acc[4]; acc[0] = vz; acc[1] = vz; acc[2] = vz; acc[3] = vz;
  for (int k = 0; k < K_; ++k) {
    // T_k^T[j][i] = p_k(L)[j][i] * Att^T[j][i]  (L symmetric -> fully contiguous)
    for (int p2 = tid; p2 < V_ * V_ / 2; p2 += 128) {
      int idx = p2 * 2;
      int j = idx >> 6, i = idx & 63;
      float l0 = sL[idx], l1 = sL[idx + 1];
      float e0 = (i == j) ? 1.f : 0.f;
      float e1 = (i + 1 == j) ? 1.f : 0.f;
      float p0 = (k == 0) ? e0 : ((k == 1) ? l0 : (2.f * l0 * l0 - e0));
      float p1 = (k == 0) ? e1 : ((k == 1) ? l1 : (2.f * l1 * l1 - e1));
      v2h h;
      h[0] = (half_t)(p0 * (float)sAttT[idx]);
      h[1] = (half_t)(p1 * (float)sAttT[idx + 1]);
      *(v2h*)(sTkT + idx) = h;
    }
    __syncthreads();
    // Z^T = X^T @ T_k : A = xa0/xa1 (hoisted), B = sTkT (cm view)
    v8f z[4]; z[0] = vz; z[1] = vz; z[2] = vz; z[3] = vz;
#pragma unroll
    for (int ni = 0; ni < 4; ++ni) {
      v16h b0 = ld_b_cm(sTkT, V_, 0, 16 * ni, lane);
      z[ni] = wmma_f16(xa0, b0, z[ni]);
      v16h b1 = ld_b_cm(sTkT, V_, 32, 16 * ni, lane);
      z[ni] = wmma_f16(xa1, b1, z[ni]);
    }
    // C = Z^T[f][j]: per-lane col j fixed, rows f contiguous in Z[j][f] -> one b128 store
#pragma unroll
    for (int ni = 0; ni < 4; ++ni) {
      int j = 16 * ni + (lane & 15), fb = 16 * wave + ((lane >> 4) << 3);
      v8h zh;
#pragma unroll
      for (int g = 0; g < 8; ++g) zh[g] = (half_t)z[ni][g];
      *(v8h*)(sZ + j * F_ + fb) = zh;
    }
    __syncthreads();
    // out^T[o][j] += Theta_k^T @ Z^T : A = ThT (rm, global), B = sZ (cm view)
#pragma unroll
    for (int ks = 0; ks < 2; ++ks) {
      v16h a = ld_a_rm(ThT + (size_t)k * CF_ * F_, F_, 16 * wave, 32 * ks, lane);
#pragma unroll
      for (int ni = 0; ni < 4; ++ni) {
        v16h b = ld_b_cm(sZ, F_, 32 * ks, 16 * ni, lane);
        acc[ni] = wmma_f16(a, b, acc[ni]);
      }
    }
    __syncthreads();
  }
  // relu + packed store: h[j][o] contiguous in o per lane
  half_t* op = hOut + (size_t)nt * V_ * CF_;
#pragma unroll
  for (int ni = 0; ni < 4; ++ni) {
    int j = 16 * ni + (lane & 15), ob = 16 * wave + ((lane >> 4) << 3);
    v8h oh;
#pragma unroll
    for (int g = 0; g < 8; ++g) { float v = acc[ni][g]; oh[g] = (half_t)(v > 0.f ? v : 0.f); }
    *(v8h*)(op + j * CF_ + ob) = oh;
  }
}

// ---------------- Chebyshev SD branch ----------------
__global__ __launch_bounds__(128) void k_cheb_sd(const float* chebpT, const float* SAtt,
    const half_t* xhT, const half_t* ThT, half_t* hOut) {
  int nt = blockIdx.x, n = nt / T_, tid = threadIdx.x;
  __shared__ __attribute__((aligned(16))) half_t sAttT[V_ * V_];
  __shared__ __attribute__((aligned(16))) half_t sTkT[V_ * V_];
  __shared__ __attribute__((aligned(16))) half_t sXT[V_ * V_];
  __shared__ __attribute__((aligned(16))) half_t sZ[V_ * V_];
  const float* Attp = SAtt + (size_t)n * V_ * V_;
  const half_t* xpT = xhT + (size_t)nt * V_ * F_;
  __builtin_prefetch(ThT, 0, 1);
  unsigned ldsX = (unsigned)(uintptr_t)sXT;
  for (int i8 = tid; i8 < V_ * F_ / 8; i8 += 128)
    async_ld_b128(ldsX + i8 * 16, (const void*)(xpT + i8 * 8));
  for (int idx = tid; idx < V_ * V_; idx += 128) {
    int i = idx >> 6, j = idx & 63;
    sAttT[j * V_ + i] = (half_t)Attp[idx];
  }
  wait_async0();
  __syncthreads();
  int wave = tid >> 5, lane = tid & 31;
  v16h xa0 = ld_a_rm(sXT, V_, 16 * wave, 0, lane);
  v16h xa1 = ld_a_rm(sXT, V_, 16 * wave, 32, lane);
  v8f vz = {};
  v8f acc[4]; acc[0] = vz; acc[1] = vz; acc[2] = vz; acc[3] = vz;
  for (int k = 0; k < K_; ++k) {
    const float* chT = chebpT + (size_t)k * V_ * V_;
    for (int p2 = tid; p2 < V_ * V_ / 2; p2 += 128) {  // (chebp_k .* Att)^T, contiguous
      int idx = p2 * 2;
      v2h h;
      h[0] = (half_t)(chT[idx] * (float)sAttT[idx]);
      h[1] = (half_t)(chT[idx + 1] * (float)sAttT[idx + 1]);
      *(v2h*)(sTkT + idx) = h;
    }
    __syncthreads();
    v8f z[4]; z[0] = vz; z[1] = vz; z[2] = vz; z[3] = vz;
#pragma unroll
    for (int ni = 0; ni < 4; ++ni) {
      v16h b0 = ld_b_cm(sTkT, V_, 0, 16 * ni, lane);
      z[ni] = wmma_f16(xa0, b0, z[ni]);
      v16h b1 = ld_b_cm(sTkT, V_, 32, 16 * ni, lane);
      z[ni] = wmma_f16(xa1, b1, z[ni]);
    }
#pragma unroll
    for (int ni = 0; ni < 4; ++ni) {
      int j = 16 * ni + (lane & 15), fb = 16 * wave + ((lane >> 4) << 3);
      v8h zh;
#pragma unroll
      for (int g = 0; g < 8; ++g) zh[g] = (half_t)z[ni][g];
      *(v8h*)(sZ + j * F_ + fb) = zh;
    }
    __syncthreads();
#pragma unroll
    for (int ks = 0; ks < 2; ++ks) {
      v16h a = ld_a_rm(ThT + (size_t)k * CF_ * F_, F_, 16 * wave, 32 * ks, lane);
#pragma unroll
      for (int ni = 0; ni < 4; ++ni) {
        v16h b = ld_b_cm(sZ, F_, 32 * ks, 16 * ni, lane);
        acc[ni] = wmma_f16(a, b, acc[ni]);
      }
    }
    __syncthreads();
  }
  half_t* op = hOut + (size_t)nt * V_ * CF_;
#pragma unroll
  for (int ni = 0; ni < 4; ++ni) {
    int j = 16 * ni + (lane & 15), ob = 16 * wave + ((lane >> 4) << 3);
    v8h oh;
#pragma unroll
    for (int g = 0; g < 8; ++g) { float v = acc[ni][g]; oh[g] = (half_t)(v > 0.f ? v : 0.f); }
    *(v8h*)(op + j * CF_ + ob) = oh;
  }
}

// ---------------- time conv (KT=3, SAME over T) + LayerNorm + concat write ----------------
// Y^T[o][v] = sum_dt W_dt^T @ H_dt^T : A = WhT (rm, global), B = sH (cm view of H^T)
__global__ __launch_bounds__(128) void k_tconv_ln(const half_t* h, const half_t* WhT,
    const float* bias, float* out, int branch) {
  int nt = blockIdx.x, n = nt / T_, t = nt % T_, tid = threadIdx.x;
  __shared__ __attribute__((aligned(16))) half_t sH[3][V_ * CF_];
  __shared__ __attribute__((aligned(16))) float sY[V_ * TF_];
  __shared__ float sBias[TF_];
  v8h zero8 = {};
  if (tid < TF_) sBias[tid] = bias[tid];
  for (int dt = 0; dt < 3; ++dt) {
    int tt = t + dt - 1;
    const v8h* src8 = (tt >= 0 && tt < T_) ? (const v8h*)(h + (size_t)(n * T_ + tt) * V_ * CF_) : nullptr;
    v8h* dst8 = (v8h*)sH[dt];
    for (int i8 = tid; i8 < V_ * CF_ / 8; i8 += 128) dst8[i8] = src8 ? src8[i8] : zero8;
  }
  __syncthreads();
  int wave = tid >> 5, lane = tid & 31;
  v8f vz = {};
  v8f acc[4]; acc[0] = vz; acc[1] = vz; acc[2] = vz; acc[3] = vz;
  for (int dt = 0; dt < 3; ++dt) {
#pragma unroll
    for (int ks = 0; ks < 2; ++ks) {
      v16h a = ld_a_rm(WhT + (size_t)dt * TF_ * CF_, CF_, 16 * wave, 32 * ks, lane);
#pragma unroll
      for (int ni = 0; ni < 4; ++ni) {
        v16h b = ld_b_cm(sH[dt], CF_, 32 * ks, 16 * ni, lane);
        acc[ni] = wmma_f16(a, b, acc[ni]);
      }
    }
  }
  // C = Y^T[o][v]: per-lane col v fixed, rows o contiguous in sY[v][o] -> 2x float4 stores
#pragma unroll
  for (int ni = 0; ni < 4; ++ni) {
    int v = 16 * ni + (lane & 15), ob = 16 * wave + ((lane >> 4) << 3);
    float4 lo, hi;
    lo.x = acc[ni][0] + sBias[ob + 0]; lo.y = acc[ni][1] + sBias[ob + 1];
    lo.z = acc[ni][2] + sBias[ob + 2]; lo.w = acc[ni][3] + sBias[ob + 3];
    hi.x = acc[ni][4] + sBias[ob + 4]; hi.y = acc[ni][5] + sBias[ob + 5];
    hi.z = acc[ni][6] + sBias[ob + 6]; hi.w = acc[ni][7] + sBias[ob + 7];
    *(float4*)(sY + v * TF_ + ob) = lo;
    *(float4*)(sY + v * TF_ + ob + 4) = hi;
  }
  __syncthreads();
  if (tid < V_) {
    int v = tid;
    float m = 0.f;
    for (int o = 0; o < TF_; ++o) m += sY[v * TF_ + o];
    m *= (1.f / TF_);
    float var = 0.f;
    for (int o = 0; o < TF_; ++o) { float d = sY[v * TF_ + o] - m; var += d * d; }
    var *= (1.f / TF_);
    float inv = rsqrtf(var + LN_EPS_);
    float* dst = out + ((size_t)nt * V_ + v) * (2 * TF_) + branch * TF_;
    for (int o = 0; o < TF_; ++o) dst[o] = (sY[v * TF_ + o] - m) * inv;
  }
}

// ---------------- host launcher ----------------
extern "C" void kernel_launch(void* const* d_in, const int* in_sizes, int n_in,
                              void* d_out, int out_size, void* d_ws, size_t ws_size,
                              hipStream_t stream) {
  (void)in_sizes; (void)n_in; (void)out_size; (void)ws_size;
  const float* x     = (const float*)d_in[0];
  const float* a_gl  = (const float*)d_in[1];
  const float* U1    = (const float*)d_in[2];
  const float* U2    = (const float*)d_in[3];
  const float* U3    = (const float*)d_in[4];
  const float* b_e   = (const float*)d_in[5];
  const float* V_e   = (const float*)d_in[6];
  const float* W1    = (const float*)d_in[7];
  const float* W2    = (const float*)d_in[8];
  const float* W3    = (const float*)d_in[9];
  const float* b_s   = (const float*)d_in[10];
  const float* V_s   = (const float*)d_in[11];
  const float* ThGL  = (const float*)d_in[12];
  const float* ThSD  = (const float*)d_in[13];
  const float* chebp = (const float*)d_in[14];
  const float* Wcgl  = (const float*)d_in[15];
  const float* bcgl  = (const float*)d_in[16];
  const float* Wcsd  = (const float*)d_in[17];
  const float* bcsd  = (const float*)d_in[18];
  float* out = (float*)d_out;

  char* ws = (char*)d_ws;
  size_t off = 0;
  auto alloc = [&](size_t bytes) -> void* {
    void* p = ws + off;
    off = (off + bytes + 255) & ~(size_t)255;
    return p;
  };
  const int NT = N_ * T_;
  const int XSZ = N_ * T_ * V_ * F_;
  float*  TAtt   = (float*)alloc(sizeof(float) * N_ * T_ * T_);
  float*  xT     = (float*)alloc(sizeof(float) * XSZ);
  float*  SAtt   = (float*)alloc(sizeof(float) * N_ * V_ * V_);
  float*  S      = (float*)alloc(sizeof(float) * N_ * T_ * V_ * V_);
  float*  Smean  = (float*)alloc(sizeof(float) * T_ * V_ * V_);
  float*  D2     = (float*)alloc(sizeof(float) * V_ * V_);
  float*  chebpT = (float*)alloc(sizeof(float) * K_ * V_ * V_);
  half_t* xhT    = (half_t*)alloc(sizeof(half_t) * XSZ);
  half_t* ThGLhT = (half_t*)alloc(sizeof(half_t) * K_ * F_ * CF_);
  half_t* ThSDhT = (half_t*)alloc(sizeof(half_t) * K_ * F_ * CF_);
  half_t* WghT   = (half_t*)alloc(sizeof(half_t) * KT_ * CF_ * TF_);
  half_t* WshT   = (half_t*)alloc(sizeof(half_t) * KT_ * CF_ * TF_);
  half_t* hGL    = (half_t*)alloc(sizeof(half_t) * N_ * T_ * V_ * CF_);
  half_t* hSD    = (half_t*)alloc(sizeof(half_t) * N_ * T_ * V_ * CF_);

  k_prep<<<48, 256, 0, stream>>>(ThGL, ThSD, Wcgl, Wcsd, chebp,
                                 ThGLhT, ThSDhT, WghT, WshT, chebpT);
  k_x2hT<<<NT, 256, 0, stream>>>(x, xhT);
  k_tatt<<<N_, 256, 0, stream>>>(x, U1, U2, U3, b_e, V_e, TAtt);
  k_xtatt<<<NT, 256, 0, stream>>>(x, TAtt, xT);
  k_satt<<<N_, 256, 0, stream>>>(xT, W1, W2, W3, b_s, V_s, SAtt);
  k_S<<<NT, 256, 0, stream>>>(x, a_gl, S);
  k_smean<<<(T_ * V_ * V_ + 255) / 256, 256, 0, stream>>>(S, Smean);
  k_d2<<<V_ * V_, 64, 0, stream>>>(x, D2);
  k_loss<<<1, 256, 0, stream>>>(Smean, D2, out, N_ * T_ * V_ * 2 * TF_);
  k_cheb_gl<<<NT, 128, 0, stream>>>(S, SAtt, xhT, ThGLhT, hGL);
  k_cheb_sd<<<NT, 128, 0, stream>>>(chebpT, SAtt, xhT, ThSDhT, hSD);
  k_tconv_ln<<<NT, 128, 0, stream>>>(hGL, WghT, bcgl, out, 0);
  k_tconv_ln<<<NT, 128, 0, stream>>>(hSD, WshT, bcsd, out, 1);
}
